// Encoder_67980742361889
// MI455X (gfx1250) — compile-verified, matrix operand order checked by reading, not run on hardware
//
#include <hip/hip_runtime.h>
#include <math.h>

typedef __attribute__((ext_vector_type(16))) _Float16 v16h;
typedef __attribute__((ext_vector_type(8)))  _Float16 v8h;
typedef __attribute__((ext_vector_type(4)))  _Float16 v4h;
typedef __attribute__((ext_vector_type(2)))  __fp16   v2hf;   // cvt_pkrtz result type
typedef __attribute__((ext_vector_type(8)))  float    v8f;
typedef __attribute__((ext_vector_type(4)))  float    v4f;

// Problem constants (match reference)
static constexpr int kN   = 3000;
static constexpr int kFIN = 3000;
static constexpr int kD   = 512;
static constexpr int kH   = 8;
static constexpr int kC   = 64;
static constexpr int kDFF = 2048;
static constexpr int kE   = 48000;
static constexpr int kNE  = kE + kN;   // edges incl. self loops

// ---------------------------------------------------------------------------
// Generic WMMA GEMM:  C[M,Nc] = act(alpha * A[M,K] * B + bias)
//   transB==0 : B is [K,Nc] row-major, leading dim ldb
//   transB==1 : B is [Nc,K] row-major, leading dim ldb  (C = A*B^T)
// Block tile 64x64, K-step 32, 8 wave32 waves; each wave does 2 WMMA tiles.
// All A/B operands passed in have lda/ldb % 4 == 0 and 16B-aligned bases, so
// the interior fast path can use b128 loads + packed f16 converts.
// ---------------------------------------------------------------------------
#define BM 64
#define BN 64
#define BK 32

__global__ __launch_bounds__(256)
void enc_gemm_wmma(const float* __restrict__ A, const float* __restrict__ B,
                   const float* __restrict__ bias, float* __restrict__ C,
                   int M, int Nc, int K, int lda, int ldb, int ldc,
                   int transB, float alpha, int act)
{
    __shared__ __align__(16) _Float16 sA[BM][BK];
    __shared__ __align__(16) _Float16 sB[BN][BK];   // stored N-major

    const int tid  = threadIdx.x;
    const int lane = tid & 31;
    const int wave = tid >> 5;      // 0..7
    const int mi   = wave & 3;      // 4 M-subtiles of 16
    const int ni   = wave >> 2;     // 2 N-groups of 32 (2 x 16-wide WMMA tiles)
    const int hsel = lane >> 4;     // lane half -> K-group select per ISA layout
    const int lrow = lane & 15;

    const int tileM = blockIdx.y * BM;
    const int tileN = blockIdx.x * BN;
    const bool interior = (tileM + BM <= M) && (tileN + BN <= Nc);

    v8f acc0 = {};
    v8f acc1 = {};

    for (int k0 = 0; k0 < K; k0 += BK) {
        // Prefetch next A K-tile (global_prefetch_b8, speculative)
        if (k0 + BK < K && tid < BM)
            __builtin_prefetch(&A[(long)(tileM + tid) * lda + k0 + BK], 0, 0);

        if (interior && (k0 + BK <= K)) {
            // ---- fast path: b128 loads, packed cvt, b64 LDS stores ----
#pragma unroll
            for (int t = 0; t < 2; ++t) {               // A: 64x32 = 512 float4
                int idx = tid + t * 256;
                int r = idx >> 3, c4 = idx & 7;
                v4f f = *(const v4f*)&A[(long)(tileM + r) * lda + k0 + c4 * 4];
                union { v4h v; v2hf h[2]; } pk;
                pk.h[0] = __builtin_amdgcn_cvt_pkrtz(f[0], f[1]);
                pk.h[1] = __builtin_amdgcn_cvt_pkrtz(f[2], f[3]);
                *(v4h*)&sA[r][c4 * 4] = pk.v;
            }
            if (transB) {
#pragma unroll
                for (int t = 0; t < 2; ++t) {           // B^T: 64 rows x 8 float4
                    int idx = tid + t * 256;
                    int n = idx >> 3, c4 = idx & 7;
                    v4f f = *(const v4f*)&B[(long)(tileN + n) * ldb + k0 + c4 * 4];
                    union { v4h v; v2hf h[2]; } pk;
                    pk.h[0] = __builtin_amdgcn_cvt_pkrtz(f[0], f[1]);
                    pk.h[1] = __builtin_amdgcn_cvt_pkrtz(f[2], f[3]);
                    *(v4h*)&sB[n][c4 * 4] = pk.v;
                }
            } else {
#pragma unroll
                for (int t = 0; t < 2; ++t) {           // B: 32 k-rows x 16 float4
                    int idx = tid + t * 256;
                    int kk = idx >> 4, n4 = idx & 15;
                    v4f f = *(const v4f*)&B[(long)(k0 + kk) * ldb + tileN + n4 * 4];
#pragma unroll
                    for (int j = 0; j < 4; ++j)
                        sB[n4 * 4 + j][kk] = (_Float16)f[j];
                }
            }
        } else {
            // ---- slow path: branchless clamped scalar loads ----
#pragma unroll
            for (int t = 0; t < 8; ++t) {               // A: 2048 elems
                int idx = tid + t * 256;
                int r = idx >> 5, c = idx & 31;
                int gm = tileM + r, gk = k0 + c;
                int cm = gm < M ? gm : M - 1;
                int ck = gk < K ? gk : K - 1;
                float v = A[(long)cm * lda + ck];
                v = (gm < M && gk < K) ? v : 0.0f;
                sA[r][c] = (_Float16)v;
            }
#pragma unroll
            for (int t = 0; t < 8; ++t) {               // B: 2048 elems, N-major
                int idx = tid + t * 256;
                int n = idx >> 5, c = idx & 31;
                int gn = tileN + n, gk = k0 + c;
                int cn = gn < Nc ? gn : Nc - 1;
                int ck = gk < K ? gk : K - 1;
                float v = transB ? B[(long)cn * ldb + ck] : B[(long)ck * ldb + cn];
                v = (gn < Nc && gk < K) ? v : 0.0f;
                sB[n][c] = (_Float16)v;
            }
        }
        __syncthreads();

        // Per-lane fragments: two ds_load_b128 each, per the 16-bit A/B VGPR
        // layout (lanes 0-15: K 0-7 & 16-23; lanes 16-31: K 8-15 & 24-31)
        union { v16h v; v8h h[2]; } af, bf0, bf1;
        af.h[0]  = *(const v8h*)&sA[mi * 16 + lrow][hsel * 8];
        af.h[1]  = *(const v8h*)&sA[mi * 16 + lrow][16 + hsel * 8];
        bf0.h[0] = *(const v8h*)&sB[ni * 32 + lrow][hsel * 8];
        bf0.h[1] = *(const v8h*)&sB[ni * 32 + lrow][16 + hsel * 8];
        bf1.h[0] = *(const v8h*)&sB[ni * 32 + 16 + lrow][hsel * 8];
        bf1.h[1] = *(const v8h*)&sB[ni * 32 + 16 + lrow][16 + hsel * 8];

        acc0 = __builtin_amdgcn_wmma_f32_16x16x32_f16(
                   false, af.v, false, bf0.v, (short)0, acc0, false, false);
        acc1 = __builtin_amdgcn_wmma_f32_16x16x32_f16(
                   false, af.v, false, bf1.v, (short)0, acc1, false, false);
        __syncthreads();
    }

    // Epilogue: C/D layout -> VGPR r holds M = r (+8 for upper lane half)
#pragma unroll
    for (int j = 0; j < 2; ++j) {
        const v8f* accp = j ? &acc1 : &acc0;
        int col = tileN + ni * 32 + j * 16 + lrow;
        if (col < Nc) {
            float bv = bias ? bias[col] : 0.0f;
#pragma unroll
            for (int r = 0; r < 8; ++r) {
                int row = tileM + mi * 16 + r + hsel * 8;
                if (row < M) {
                    float v = (*accp)[r] * alpha + bv;
                    if (act == 1)      v = v > 0.0f ? v : 0.0f;        // relu
                    else if (act == 2) v = v > 0.0f ? v : 0.01f * v;   // leaky 0.01
                    C[(long)row * ldc + col] = v;
                }
            }
        }
    }
}

// ---------------------------------------------------------------------------
// Small helper kernels
// ---------------------------------------------------------------------------
__global__ void enc_fill_f32(float* p, float v, long n) {
    long i = (long)blockIdx.x * blockDim.x + threadIdx.x;
    if (i < n) p[i] = v;
}

__global__ void enc_build_edges(const int* __restrict__ ei, int* src, int* dst) {
    int i = blockIdx.x * blockDim.x + threadIdx.x;
    if (i < kE)        { src[i] = ei[i]; dst[i] = ei[kE + i]; }
    else if (i < kNE)  { src[i] = i - kE; dst[i] = i - kE; }
}

__device__ inline void enc_atomicMaxF(float* addr, float val) {
    if (val >= 0.0f) atomicMax((int*)addr, __float_as_int(val));
    else             atomicMin((unsigned int*)addr, __float_as_uint(val));
}

// Per (edge, head): logit = att_h . leaky_relu(xl[src] + xr[dst], 0.2)
__global__ void enc_gat_logits(const float* __restrict__ xl, const float* __restrict__ xr,
                               const float* __restrict__ att,
                               const int* __restrict__ src, const int* __restrict__ dst,
                               float* __restrict__ logit)
{
    int i = blockIdx.x * blockDim.x + threadIdx.x;
    if (i >= kNE * kH) return;
    int e = i >> 3, h = i & 7;
    const float* pl = xl + (long)src[e] * kD + h * kC;
    const float* pr = xr + (long)dst[e] * kD + h * kC;
    const float* pa = att + h * kC;
    float acc = 0.0f;
#pragma unroll 4
    for (int c = 0; c < kC; ++c) {
        float v = pl[c] + pr[c];
        v = v > 0.0f ? v : 0.2f * v;
        acc += pa[c] * v;
    }
    logit[(long)e * kH + h] = acc;
}

__global__ void enc_seg_max(const float* __restrict__ logit, const int* __restrict__ dst,
                            float* __restrict__ mx)
{
    int i = blockIdx.x * blockDim.x + threadIdx.x;
    if (i >= kNE * kH) return;
    int e = i >> 3, h = i & 7;
    enc_atomicMaxF(&mx[dst[e] * kH + h], logit[i]);
}

__global__ void enc_seg_expsum(float* __restrict__ logit, const int* __restrict__ dst,
                               const float* __restrict__ mx, float* __restrict__ sm)
{
    int i = blockIdx.x * blockDim.x + threadIdx.x;
    if (i >= kNE * kH) return;
    int e = i >> 3, h = i & 7;
    float ev = expf(logit[i] - mx[dst[e] * kH + h]);
    logit[i] = ev;
    atomicAdd(&sm[dst[e] * kH + h], ev);
}

// Per (edge, feature): g[dst] += alpha * xl[src]
__global__ void enc_gat_scatter(const float* __restrict__ ebuf, const float* __restrict__ sm,
                                const float* __restrict__ xl,
                                const int* __restrict__ src, const int* __restrict__ dst,
                                float* __restrict__ g)
{
    long i = (long)blockIdx.x * blockDim.x + threadIdx.x;
    if (i >= (long)kNE * kD) return;
    int e = (int)(i >> 9);
    int d = (int)(i & (kD - 1));
    int h = d >> 6;
    float a = ebuf[(long)e * kH + h] / sm[(long)dst[e] * kH + h];
    atomicAdd(&g[(long)dst[e] * kD + d], a * xl[(long)src[e] * kD + d]);
}

// Row softmax; row staged in LDS (1 global read + 1 write instead of 3R+2W)
__global__ __launch_bounds__(256)
void enc_softmax_rows(float* __restrict__ x, int L)
{
    __shared__ float red[256];
    __shared__ float srow[3072];     // L <= 3000 here
    float* row = x + (long)blockIdx.x * L;
    int tid = threadIdx.x;
    float m = -3.4e38f;
    for (int j = tid; j < L; j += 256) { float v = row[j]; srow[j] = v; m = fmaxf(m, v); }
    red[tid] = m; __syncthreads();
    for (int s = 128; s > 0; s >>= 1) { if (tid < s) red[tid] = fmaxf(red[tid], red[tid + s]); __syncthreads(); }
    m = red[0]; __syncthreads();
    float sum = 0.0f;
    for (int j = tid; j < L; j += 256) { float e = expf(srow[j] - m); srow[j] = e; sum += e; }
    red[tid] = sum; __syncthreads();
    for (int s = 128; s > 0; s >>= 1) { if (tid < s) red[tid] += red[tid + s]; __syncthreads(); }
    float inv = 1.0f / red[0];
    __syncthreads();
    for (int j = tid; j < L; j += 256) row[j] = srow[j] * inv;
}

// x = LayerNorm(x + y) * g + b ; row (Dd<=512) staged in LDS
__global__ __launch_bounds__(256)
void enc_ln_residual(float* __restrict__ x, const float* __restrict__ y,
                     const float* __restrict__ g, const float* __restrict__ b, int Dd)
{
    __shared__ float red[256];
    __shared__ float buf[512];
    long base = (long)blockIdx.x * Dd;
    int tid = threadIdx.x;
    float s = 0.0f;
    for (int j = tid; j < Dd; j += 256) { float v = x[base + j] + y[base + j]; buf[j] = v; s += v; }
    red[tid] = s; __syncthreads();
    for (int st = 128; st > 0; st >>= 1) { if (tid < st) red[tid] += red[tid + st]; __syncthreads(); }
    float mean = red[0] / (float)Dd; __syncthreads();
    float vs = 0.0f;
    for (int j = tid; j < Dd; j += 256) { float v = buf[j] - mean; vs += v * v; }
    red[tid] = vs; __syncthreads();
    for (int st = 128; st > 0; st >>= 1) { if (tid < st) red[tid] += red[tid + st]; __syncthreads(); }
    float inv = rsqrtf(red[0] / (float)Dd + 1e-5f);
    for (int j = tid; j < Dd; j += 256)
        x[base + j] = (buf[j] - mean) * inv * g[j] + b[j];
}

// comb = a*(g+gb) + (1-a)*t ; emb = leaky_relu(comb, 0.01)
__global__ void enc_combine(const float* __restrict__ g, const float* __restrict__ gb,
                            const float* __restrict__ t, const float* __restrict__ alphap,
                            float* __restrict__ comb, float* __restrict__ emb, long n)
{
    long i = (long)blockIdx.x * blockDim.x + threadIdx.x;
    if (i >= n) return;
    float a = 1.0f / (1.0f + expf(-alphap[0]));
    float cv = a * (g[i] + gb[i & (kD - 1)]) + (1.0f - a) * t[i];
    comb[i] = cv;
    emb[i]  = cv > 0.0f ? cv : 0.01f * cv;
}

__global__ __launch_bounds__(256)
void enc_rowsum(const float* __restrict__ A, float* __restrict__ s, int L)
{
    __shared__ float red[256];
    const float* row = A + (long)blockIdx.x * L;
    float acc = 0.0f;
    for (int j = threadIdx.x; j < L; j += 256) acc += row[j];
    red[threadIdx.x] = acc; __syncthreads();
    for (int st = 128; st > 0; st >>= 1) { if (threadIdx.x < st) red[threadIdx.x] += red[threadIdx.x + st]; __syncthreads(); }
    if (threadIdx.x == 0) s[blockIdx.x] = red[0];
}

// c = s/msum; c /= max(||c||,1e-12); ctx = sigmoid(c)  (block per row)
__global__ __launch_bounds__(256)
void enc_readout_finish(const float* __restrict__ s, const float* __restrict__ msum,
                        float* __restrict__ ctx, int Dd)
{
    __shared__ float red[256];
    __shared__ float buf[512];
    long base = (long)blockIdx.x * Dd;
    float inv = 1.0f / msum[blockIdx.x];
    float acc = 0.0f;
    for (int j = threadIdx.x; j < Dd; j += 256) { float c = s[base + j] * inv; buf[j] = c; acc += c * c; }
    red[threadIdx.x] = acc; __syncthreads();
    for (int st = 128; st > 0; st >>= 1) { if (threadIdx.x < st) red[threadIdx.x] += red[threadIdx.x + st]; __syncthreads(); }
    float innorm = 1.0f / fmaxf(sqrtf(red[0]), 1e-12f);
    for (int j = threadIdx.x; j < Dd; j += 256) {
        float c = buf[j] * innorm;
        ctx[base + j] = 1.0f / (1.0f + expf(-c));
    }
}

// logits[n] = [dot(pos_n, u_n)+bb, dot(neg_n, u_n)+bb]
__global__ void enc_disc(const float* __restrict__ pos, const float* __restrict__ neg,
                         const float* __restrict__ u, const float* __restrict__ bb,
                         float* __restrict__ out2)
{
    int i = blockIdx.x * blockDim.x + threadIdx.x;
    if (i >= kN) return;
    const float* pu = u + (long)i * kD;
    const float* pp = pos + (long)i * kD;
    const float* pn = neg + (long)i * kD;
    float s1 = 0.0f, s2 = 0.0f;
    for (int j = 0; j < kD; ++j) { s1 += pp[j] * pu[j]; s2 += pn[j] * pu[j]; }
    out2[i * 2]     = s1 + bb[0];
    out2[i * 2 + 1] = s2 + bb[0];
}

// ---------------------------------------------------------------------------
// Host orchestration
// ---------------------------------------------------------------------------
extern "C" void kernel_launch(void* const* d_in, const int* in_sizes, int n_in,
                              void* d_out, int out_size, void* d_ws, size_t ws_size,
                              hipStream_t stream)
{
    (void)in_sizes; (void)n_in; (void)out_size; (void)ws_size;

    const float* feat   = (const float*)d_in[0];
    const float* feat_a = (const float*)d_in[1];
    const int*   eidx   = (const int*)d_in[2];
    const float* adj    = (const float*)d_in[3];
    const float* Wl     = (const float*)d_in[4];
    const float* Wr     = (const float*)d_in[5];
    const float* att    = (const float*)d_in[6];
    const float* gat_b  = (const float*)d_in[7];
    const float* Wproj  = (const float*)d_in[8];
    const float* bproj  = (const float*)d_in[9];
    const float* Wqkv   = (const float*)d_in[10];
    const float* bqkv   = (const float*)d_in[11];
    const float* Wo     = (const float*)d_in[12];
    const float* bo     = (const float*)d_in[13];
    const float* ln1_g  = (const float*)d_in[14];
    const float* ln1_b  = (const float*)d_in[15];
    const float* ln2_g  = (const float*)d_in[16];
    const float* ln2_b  = (const float*)d_in[17];
    const float* Wff1   = (const float*)d_in[18];
    const float* bff1   = (const float*)d_in[19];
    const float* Wff2   = (const float*)d_in[20];
    const float* bff2   = (const float*)d_in[21];
    const float* alphap = (const float*)d_in[22];
    const float* Wout   = (const float*)d_in[23];
    const float* bout   = (const float*)d_in[24];
    const float* Wdisc  = (const float*)d_in[25];
    const float* bdisc  = (const float*)d_in[26];

    float* out      = (float*)d_out;
    float* emb0     = out;                          // [N, D]
    float* decoded  = out + (long)kN * kD;          // [N, FIN]
    float* logits   = decoded + (long)kN * kFIN;    // [N, 2]
    float* logits_a = logits + kN * 2;              // [N, 2]

    // Workspace carve-out (256B aligned blocks)
    char* wp = (char*)d_ws;
    auto alloc = [&](size_t bytes) {
        void* p = (void*)wp;
        wp += (bytes + 255) & ~(size_t)255;
        return p;
    };
    int*   srcb   = (int*)  alloc((size_t)kNE * sizeof(int));
    int*   dstb   = (int*)  alloc((size_t)kNE * sizeof(int));
    float* xl     = (float*)alloc((size_t)kN * kD * 4);
    float* xr     = (float*)alloc((size_t)kN * kD * 4);
    float* elog   = (float*)alloc((size_t)kNE * kH * 4);
    float* segmax = (float*)alloc((size_t)kN * kH * 4);
    float* segsum = (float*)alloc((size_t)kN * kH * 4);
    float* gbuf   = (float*)alloc((size_t)kN * kD * 4);
    float* tx     = (float*)alloc((size_t)kN * kD * 4);
    float* qkvb   = (float*)alloc((size_t)kN * 3 * kD * 4);
    float* sbuf   = (float*)alloc((size_t)kN * kN * 4);
    float* obuf   = (float*)alloc((size_t)kN * kD * 4);
    float* pbuf   = (float*)alloc((size_t)kN * kD * 4);
    float* fbuf   = (float*)alloc((size_t)kN * kDFF * 4);
    float* cbuf   = (float*)alloc((size_t)kN * kD * 4);
    float* emba   = (float*)alloc((size_t)kN * kD * 4);
    float* msum   = (float*)alloc((size_t)kN * 4);
    float* ctx    = (float*)alloc((size_t)kN * kD * 4);
    float* ubuf   = (float*)alloc((size_t)kN * kD * 4);

    auto GEMM = [&](const float* A, const float* B, const float* bias, float* Cc,
                    int M, int Nc, int K, int lda, int ldb, int ldc,
                    int transB, float alpha, int act) {
        dim3 grid((Nc + BN - 1) / BN, (M + BM - 1) / BM);
        enc_gemm_wmma<<<grid, dim3(256), 0, stream>>>(A, B, bias, Cc, M, Nc, K,
                                                      lda, ldb, ldc, transB, alpha, act);
    };

    // ---- shared precompute ----
    enc_build_edges<<<(kNE + 255) / 256, 256, 0, stream>>>(eidx, srcb, dstb);
    enc_rowsum<<<kN, 256, 0, stream>>>(adj, msum, kN);

    auto run_branch = [&](const float* x, float* emb, float* comb) {
        // ---- GATv2 ----
        GEMM(x, Wl, nullptr, xl, kN, kD, kFIN, kFIN, kD, kD, 0, 1.0f, 0);
        GEMM(x, Wr, nullptr, xr, kN, kD, kFIN, kFIN, kD, kD, 0, 1.0f, 0);
        enc_gat_logits<<<(kNE * kH + 255) / 256, 256, 0, stream>>>(xl, xr, att, srcb, dstb, elog);
        enc_fill_f32<<<(kN * kH + 255) / 256, 256, 0, stream>>>(segmax, -1e30f, (long)kN * kH);
        enc_fill_f32<<<(kN * kH + 255) / 256, 256, 0, stream>>>(segsum, 0.0f, (long)kN * kH);
        enc_seg_max<<<(kNE * kH + 255) / 256, 256, 0, stream>>>(elog, dstb, segmax);
        enc_seg_expsum<<<(kNE * kH + 255) / 256, 256, 0, stream>>>(elog, dstb, segmax, segsum);
        enc_fill_f32<<<(int)(((long)kN * kD + 255) / 256), 256, 0, stream>>>(gbuf, 0.0f, (long)kN * kD);
        enc_gat_scatter<<<(int)(((long)kNE * kD + 255) / 256), 256, 0, stream>>>(
            elog, segsum, xl, srcb, dstb, gbuf);

        // ---- transformer ----
        GEMM(x, Wproj, bproj, tx, kN, kD, kFIN, kFIN, kD, kD, 0, 1.0f, 0);
        for (int l = 0; l < 2; ++l) {
            GEMM(tx, Wqkv + (long)l * 3 * kD * kD, bqkv + l * 3 * kD, qkvb,
                 kN, 3 * kD, kD, kD, kD, 3 * kD, 1, 1.0f, 0);
            for (int h = 0; h < kH; ++h) {
                // scores = (Q_h K_h^T) / 8
                GEMM(qkvb + h * kC, qkvb + kD + h * kC, nullptr, sbuf,
                     kN, kN, kC, 3 * kD, 3 * kD, kN, 1, 0.125f, 0);
                enc_softmax_rows<<<kN, 256, 0, stream>>>(sbuf, kN);
                // o_h = scores @ V_h
                GEMM(sbuf, qkvb + 2 * kD + h * kC, nullptr, obuf + h * kC,
                     kN, kC, kN, kN, 3 * kD, kD, 0, 1.0f, 0);
            }
            GEMM(obuf, Wo + (long)l * kD * kD, bo + l * kD, pbuf,
                 kN, kD, kD, kD, kD, kD, 1, 1.0f, 0);
            enc_ln_residual<<<kN, 256, 0, stream>>>(tx, pbuf, ln1_g + l * kD, ln1_b + l * kD, kD);
            GEMM(tx, Wff1 + (long)l * kD * kDFF, bff1 + l * kDFF, fbuf,
                 kN, kDFF, kD, kD, kDFF, kDFF, 0, 1.0f, 1);
            GEMM(fbuf, Wff2 + (long)l * kDFF * kD, bff2 + l * kD, pbuf,
                 kN, kD, kDFF, kDFF, kD, kD, 0, 1.0f, 0);
            enc_ln_residual<<<kN, 256, 0, stream>>>(tx, pbuf, ln2_g + l * kD, ln2_b + l * kD, kD);
        }

        enc_combine<<<(int)(((long)kN * kD + 255) / 256), 256, 0, stream>>>(
            gbuf, gat_b, tx, alphap, comb, emb, (long)kN * kD);
    };

    run_branch(feat,   emb0, cbuf);   // keep combined for decoder
    run_branch(feat_a, emba, pbuf);   // combined for aug branch unused

    // ---- decoder ----
    GEMM(cbuf, Wout, bout, decoded, kN, kFIN, kD, kD, kFIN, kFIN, 0, 1.0f, 0);

    // ---- readout + discriminator (branch 0) ----
    GEMM(adj, emb0, nullptr, sbuf, kN, kD, kN, kN, kD, kD, 0, 1.0f, 0);
    enc_readout_finish<<<kN, 256, 0, stream>>>(sbuf, msum, ctx, kD);
    GEMM(ctx, Wdisc, nullptr, ubuf, kN, kD, kD, kD, kD, kD, 1, 1.0f, 0);
    enc_disc<<<(kN + 255) / 256, 256, 0, stream>>>(emb0, emba, ubuf, bdisc, logits);

    // ---- readout + discriminator (branch a) ----
    GEMM(adj, emba, nullptr, sbuf, kN, kD, kN, kN, kD, kD, 0, 1.0f, 0);
    enc_readout_finish<<<kN, 256, 0, stream>>>(sbuf, msum, ctx, kD);
    GEMM(ctx, Wdisc, nullptr, ubuf, kN, kD, kD, kD, kD, kD, 1, 1.0f, 0);
    enc_disc<<<(kN + 255) / 256, 256, 0, stream>>>(emba, emb0, ubuf, bdisc, logits_a);
}
